// EquivariantRnn_6133213299291
// MI455X (gfx1250) — matrix operand order repaired and checked by reading, not compiled, not run on hardware
//
#include <hip/hip_runtime.h>
#include <math.h>

#define HID   512
#define NEMB  512
#define IDX9  9
#define BB    16
#define TT    64
#define SEQN  (IDX9*BB*TT)   // 9216
#define BT    (BB*TT)        // 1024
#define KFIN  (IDX9*HID)     // 4608

typedef __bf16 bf16_t;
typedef __attribute__((ext_vector_type(16))) __bf16 v16bf;
typedef __attribute__((ext_vector_type(8)))  __bf16 v8bf;
typedef __attribute__((ext_vector_type(8)))  float  v8f;

__device__ __forceinline__ unsigned short f2bf_bits(float x) {
  unsigned int u = __float_as_uint(x);
  unsigned int r = u + 0x7FFFu + ((u >> 16) & 1u);   // round-to-nearest-even
  return (unsigned short)(r >> 16);
}

__device__ __forceinline__ v16bf cat16(v8bf lo, v8bf hi) {
  return __builtin_shufflevector(lo, hi, 0,1,2,3,4,5,6,7,8,9,10,11,12,13,14,15);
}

// WMMA A fragment (16x32 bf16) from row-major W[*, ld] at (row0, k0).
// Layout: lane m (0..15)/hi=0 holds K=0..15; hi=1 holds K=16..31.
__device__ __forceinline__ v16bf load_a_frag(const bf16_t* W, int row0, int k0, int ld) {
  int lane = threadIdx.x & 31;
  int m = lane & 15, hi = lane >> 4;
  const bf16_t* p = W + (size_t)(row0 + m) * ld + k0 + hi * 16;
  v8bf lo = *(const v8bf*)(p);
  v8bf hv = *(const v8bf*)(p + 8);
  return cat16(lo, hv);
}

// WMMA B fragment (32x16 bf16) where logical B[k][n] = Bt[n0+n][k0+k], Bt row-major.
// Contiguous K order, same as load_a_frag: lanes 0-15 hold K=0..15, lanes 16-31 hold K=16..31 (N = lane&15).
__device__ __forceinline__ v16bf load_bt_frag(const bf16_t* Bt, int n0, int k0, int ld) {
  int lane = threadIdx.x & 31;
  int n = lane & 15, hi = lane >> 4;
  const bf16_t* p = Bt + (size_t)(n0 + n) * ld + k0 + hi * 16;
  v8bf lo = *(const v8bf*)(p);
  v8bf hv = *(const v8bf*)(p + 8);
  return cat16(lo, hv);
}

// Broadcast-vector B fragment: B[k][n] = v[k0+k] for every column n (LDS source).
__device__ __forceinline__ v16bf load_vec_frag(const unsigned short* v, int k0) {
  int hi = (threadIdx.x & 31) >> 4;
  const bf16_t* p = (const bf16_t*)(v + k0 + hi * 16);
  v8bf lo = *(const v8bf*)(p);
  v8bf hv = *(const v8bf*)(p + 8);
  return cat16(lo, hv);
}

__device__ __forceinline__ v8f wmma_bf16(v16bf a, v16bf b, v8f c) {
  return __builtin_amdgcn_wmma_f32_16x16x32_bf16(false, a, false, b, (short)0, c, false, false);
}

// ---------------- prep kernels ----------------

__global__ void k_f32_to_bf16(const float* __restrict__ src,
                              unsigned short* __restrict__ dst, int n) {
  int i = blockIdx.x * blockDim.x + threadIdx.x;
  if (i < n) dst[i] = f2bf_bits(src[i]);
}

// dst[j*HID + k] = bf16(W_ad[k*NEMB + j])  (transpose 512x512)
__global__ void k_transpose_bf16(const float* __restrict__ src,
                                 unsigned short* __restrict__ dst) {
  int i = blockIdx.x * blockDim.x + threadIdx.x;
  if (i >= HID * NEMB) return;
  int j = i >> 9, k = i & 511;
  dst[(size_t)j * HID + k] = f2bf_bits(src[(size_t)k * NEMB + j]);
}

// c = Wih0@b_ad + bih0 + bhh0 ; b1 = bih1 + bhh1
__global__ void k_vectors(const float* __restrict__ Wih0, const float* __restrict__ b_ad,
                          const float* __restrict__ bih0, const float* __restrict__ bhh0,
                          const float* __restrict__ bih1, const float* __restrict__ bhh1,
                          float* __restrict__ cvec, float* __restrict__ b1vec) {
  int i = blockIdx.x * blockDim.x + threadIdx.x;
  if (i >= HID) return;
  float s = bih0[i] + bhh0[i];
  for (int k = 0; k < HID; ++k) s += Wih0[(size_t)i * HID + k] * b_ad[k];
  cvec[i] = s;
  b1vec[i] = bih1[i] + bhh1[i];
}

// M = Wih0 @ W_ad  (512x512x512).
// Each wave: 1 row-tile x 4 col-tiles (A-fragment reused 4x per K-step).
__global__ void k_gemm_M(const unsigned short* __restrict__ Wih0_bf,
                         const unsigned short* __restrict__ WadT_bf,
                         float* __restrict__ Mout) {
  int wave = threadIdx.x >> 5;
  int job  = blockIdx.x * 8 + wave;              // 256 jobs: 32 row-tiles x 8 col-groups
  int row0 = (job >> 3) * 16;
  int col0 = (job & 7) * 64;
  const bf16_t* A  = (const bf16_t*)Wih0_bf;
  const bf16_t* Bt = (const bf16_t*)WadT_bf;
  v8f acc0 = {}, acc1 = {}, acc2 = {}, acc3 = {};
  for (int k = 0; k < HID; k += 32) {
    v16bf a  = load_a_frag(A, row0, k, HID);
    v16bf b0 = load_bt_frag(Bt, col0,      k, HID);
    v16bf b1 = load_bt_frag(Bt, col0 + 16, k, HID);
    v16bf b2 = load_bt_frag(Bt, col0 + 32, k, HID);
    v16bf b3 = load_bt_frag(Bt, col0 + 48, k, HID);
    acc0 = wmma_bf16(a, b0, acc0);
    acc1 = wmma_bf16(a, b1, acc1);
    acc2 = wmma_bf16(a, b2, acc2);
    acc3 = wmma_bf16(a, b3, acc3);
  }
  int lane = threadIdx.x & 31;
  int n = lane & 15, hi = lane >> 4;
  #pragma unroll
  for (int j = 0; j < 8; ++j) {
    size_t r = (size_t)(row0 + hi * 8 + j) * HID;
    Mout[r + col0      + n] = acc0[j];
    Mout[r + col0 + 16 + n] = acc1[j];
    Mout[r + col0 + 32 + n] = acc2[j];
    Mout[r + col0 + 48 + n] = acc3[j];
  }
}

// raw_emb[b,t,h] = b_ly2[h] + sum_n W_ly2[h, n*512 + seq[b,t,n]]  -> staged in d_out
__global__ void k_raw_emb(const int* __restrict__ seq, const float* __restrict__ W_ly2,
                          const float* __restrict__ b_ly2, float* __restrict__ out) {
  __shared__ int sIdx[IDX9];
  int bt = blockIdx.x;
  if (threadIdx.x < IDX9) sIdx[threadIdx.x] = seq[bt * IDX9 + threadIdx.x];
  __syncthreads();
  int h = threadIdx.x;
  float s = b_ly2[h];
  #pragma unroll
  for (int n = 0; n < IDX9; ++n)
    s += W_ly2[(size_t)h * (IDX9 * NEMB) + n * NEMB + sIdx[n]];
  out[(size_t)bt * HID + h] = s;
}

// ---------------- serial RNN scan (1 workgroup, 16 waves) ----------------
// step s = n*1024 + b*64 + t :
//   h0 = tanh(M[:,idx] + c + Whh0@h0) ; h1 = tanh(Wih1@h0 + Whh1@h1 + b1)
// Each wave owns 32 output rows (two 16-row WMMA tiles); h broadcast as B operand.
__global__ void k_scan(const int* __restrict__ seq, const float* __restrict__ Mmat,
                       const float* __restrict__ cvec, const float* __restrict__ b1vec,
                       const unsigned short* __restrict__ Whh0_bf,
                       const unsigned short* __restrict__ Wih1_bf,
                       const unsigned short* __restrict__ Whh1_bf,
                       const float* __restrict__ h_init,
                       unsigned short* __restrict__ hidden_bf) {
  __shared__ __align__(16) unsigned short h0b[HID];
  __shared__ __align__(16) unsigned short h1b[HID];
  __shared__ __align__(16) float u0[HID];
  __shared__ __align__(16) float u1[HID];

  const int tid = threadIdx.x;          // 512 threads
  const int wave = tid >> 5;
  const int lane = tid & 31;
  const int row0 = wave * 32;

  const bf16_t* A0 = (const bf16_t*)Whh0_bf;
  const bf16_t* A1 = (const bf16_t*)Wih1_bf;
  const bf16_t* A2 = (const bf16_t*)Whh1_bf;

  h0b[tid] = f2bf_bits(h_init[tid]);
  h1b[tid] = f2bf_bits(h_init[HID + tid]);
  __syncthreads();

  for (int s = 0; s < SEQN; ++s) {
    int n = s >> 10;
    int b = (s >> 6) & 15;
    int t = s & 63;
    int idx = seq[(b * TT + t) * IDX9 + n];
    float xreg = Mmat[(size_t)tid * NEMB + idx] + cvec[tid];
    if (s + 1 < SEQN) {
      int s2 = s + 1;
      __builtin_prefetch(&seq[(((s2 >> 6) & 15) * TT + (s2 & 63)) * IDX9 + (s2 >> 10)], 0, 0);
    }

    // u0 = Whh0 @ h0_prev
    v8f acc0 = {}, acc1 = {};
    for (int k = 0; k < HID; k += 32) {
      v16bf bv = load_vec_frag(h0b, k);
      v16bf a0 = load_a_frag(A0, row0,      k, HID);
      v16bf a1 = load_a_frag(A0, row0 + 16, k, HID);
      acc0 = wmma_bf16(a0, bv, acc0);
      acc1 = wmma_bf16(a1, bv, acc1);
    }
    if ((lane & 15) == 0) {             // lanes 0,16 hold column 0 (rows hi*8..hi*8+7)
      int hi = lane >> 4;
      #pragma unroll
      for (int j = 0; j < 8; ++j) {
        u0[row0 +      hi * 8 + j] = acc0[j];
        u0[row0 + 16 + hi * 8 + j] = acc1[j];
      }
    }
    __syncthreads();

    float v0 = tanhf(xreg + u0[tid]);
    h0b[tid] = f2bf_bits(v0);
    __syncthreads();

    // u1 = Wih1 @ h0 + Whh1 @ h1_prev
    v8f acc2 = {}, acc3 = {};
    for (int k = 0; k < HID; k += 32) {
      v16bf bv0 = load_vec_frag(h0b, k);
      v16bf a0 = load_a_frag(A1, row0,      k, HID);
      v16bf a1 = load_a_frag(A1, row0 + 16, k, HID);
      acc2 = wmma_bf16(a0, bv0, acc2);
      acc3 = wmma_bf16(a1, bv0, acc3);
      v16bf bv1 = load_vec_frag(h1b, k);
      v16bf a2 = load_a_frag(A2, row0,      k, HID);
      v16bf a3 = load_a_frag(A2, row0 + 16, k, HID);
      acc2 = wmma_bf16(a2, bv1, acc2);
      acc3 = wmma_bf16(a3, bv1, acc3);
    }
    if ((lane & 15) == 0) {
      int hi = lane >> 4;
      #pragma unroll
      for (int j = 0; j < 8; ++j) {
        u1[row0 +      hi * 8 + j] = acc2[j];
        u1[row0 + 16 + hi * 8 + j] = acc3[j];
      }
    }
    __syncthreads();

    float v1 = tanhf(u1[tid] + b1vec[tid]);
    unsigned short v1b = f2bf_bits(v1);
    h1b[tid] = v1b;
    hidden_bf[(size_t)((b * TT + t) * IDX9 + n) * HID + tid] = v1b;  // [B,T,9,H]
    __syncthreads();
  }
}

// ---------------- feat GEMM + fused epilogue ----------------
// feat[r,n] = sum_k hidden[r,k]*W_fin[n,k] + b_fin[n];  out = raw * (1 + relu(feat))
// Each wave: 1 row-tile x 4 col-tiles (A-fragment reused 4x per K-step).
__global__ void k_feat_final(const unsigned short* __restrict__ hidden_bf,
                             const unsigned short* __restrict__ Wfin_bf,
                             const float* __restrict__ b_fin, float* __restrict__ out) {
  int wave = threadIdx.x >> 5;
  int job  = blockIdx.x * 8 + wave;              // 512 jobs: 64 row-tiles x 8 col-groups
  int row0 = (job >> 3) * 16;
  int col0 = (job & 7) * 64;
  const bf16_t* A  = (const bf16_t*)hidden_bf;   // [1024][4608] row-major
  const bf16_t* Bt = (const bf16_t*)Wfin_bf;     // [512][4608]  row-major (= B transposed)
  v8f acc0 = {}, acc1 = {}, acc2 = {}, acc3 = {};
  for (int k = 0; k < KFIN; k += 32) {
    v16bf a  = load_a_frag(A, row0, k, KFIN);
    v16bf b0 = load_bt_frag(Bt, col0,      k, KFIN);
    v16bf b1 = load_bt_frag(Bt, col0 + 16, k, KFIN);
    v16bf b2 = load_bt_frag(Bt, col0 + 32, k, KFIN);
    v16bf b3 = load_bt_frag(Bt, col0 + 48, k, KFIN);
    acc0 = wmma_bf16(a, b0, acc0);
    acc1 = wmma_bf16(a, b1, acc1);
    acc2 = wmma_bf16(a, b2, acc2);
    acc3 = wmma_bf16(a, b3, acc3);
  }
  int lane = threadIdx.x & 31;
  int n = lane & 15, hi = lane >> 4;
  #pragma unroll
  for (int j = 0; j < 8; ++j) {
    int r = row0 + hi * 8 + j;
    #pragma unroll
    for (int tcol = 0; tcol < 4; ++tcol) {
      int c = col0 + tcol * 16 + n;
      float f = (tcol == 0 ? acc0[j] : tcol == 1 ? acc1[j] : tcol == 2 ? acc2[j] : acc3[j])
                + b_fin[c];
      float relu = f > 0.f ? f : 0.f;
      size_t o = (size_t)r * HID + c;
      out[o] = out[o] * (1.f + relu);            // raw staged in d_out by k_raw_emb
    }
  }
}

// ---------------- launch ----------------

extern "C" void kernel_launch(void* const* d_in, const int* in_sizes, int n_in,
                              void* d_out, int out_size, void* d_ws, size_t ws_size,
                              hipStream_t stream) {
  (void)in_sizes; (void)n_in; (void)out_size; (void)ws_size;
  const int*   seq    = (const int*)d_in[0];
  const float* W_ad   = (const float*)d_in[1];
  const float* b_ad   = (const float*)d_in[2];
  const float* W_ly2  = (const float*)d_in[3];
  const float* b_ly2  = (const float*)d_in[4];
  const float* W_fin  = (const float*)d_in[5];
  const float* b_fin  = (const float*)d_in[6];
  const float* Wih0   = (const float*)d_in[7];
  const float* Whh0   = (const float*)d_in[8];
  const float* bih0   = (const float*)d_in[9];
  const float* bhh0   = (const float*)d_in[10];
  const float* Wih1   = (const float*)d_in[11];
  const float* Whh1   = (const float*)d_in[12];
  const float* bih1   = (const float*)d_in[13];
  const float* bhh1   = (const float*)d_in[14];
  const float* h_init = (const float*)d_in[15];
  float* out = (float*)d_out;

  char* ws = (char*)d_ws;
  size_t off = 0;
  auto alloc = [&](size_t bytes) -> void* {
    void* p = ws + off;
    off = (off + bytes + 255) & ~(size_t)255;
    return p;
  };
  unsigned short* Wih0_bf  = (unsigned short*)alloc((size_t)HID * HID * 2);
  unsigned short* WadT_bf  = (unsigned short*)alloc((size_t)HID * NEMB * 2);
  unsigned short* Whh0_bf  = (unsigned short*)alloc((size_t)HID * HID * 2);
  unsigned short* Wih1_bf  = (unsigned short*)alloc((size_t)HID * HID * 2);
  unsigned short* Whh1_bf  = (unsigned short*)alloc((size_t)HID * HID * 2);
  unsigned short* Wfin_bf  = (unsigned short*)alloc((size_t)HID * KFIN * 2);
  float* Mmat   = (float*)alloc((size_t)HID * NEMB * 4);
  float* cvec   = (float*)alloc((size_t)HID * 4);
  float* b1vec  = (float*)alloc((size_t)HID * 4);
  unsigned short* hidden_bf = (unsigned short*)alloc((size_t)SEQN * HID * 2);

  const int nsq = HID * HID;
  const int nfin = HID * KFIN;
  k_f32_to_bf16<<<(nsq + 255) / 256, 256, 0, stream>>>(Wih0, Wih0_bf, nsq);
  k_f32_to_bf16<<<(nsq + 255) / 256, 256, 0, stream>>>(Whh0, Whh0_bf, nsq);
  k_f32_to_bf16<<<(nsq + 255) / 256, 256, 0, stream>>>(Wih1, Wih1_bf, nsq);
  k_f32_to_bf16<<<(nsq + 255) / 256, 256, 0, stream>>>(Whh1, Whh1_bf, nsq);
  k_f32_to_bf16<<<(nfin + 255) / 256, 256, 0, stream>>>(W_fin, Wfin_bf, nfin);
  k_transpose_bf16<<<(nsq + 255) / 256, 256, 0, stream>>>(W_ad, WadT_bf);
  k_vectors<<<2, 256, 0, stream>>>(Wih0, b_ad, bih0, bhh0, bih1, bhh1, cvec, b1vec);
  k_gemm_M<<<32, 256, 0, stream>>>(Wih0_bf, WadT_bf, Mmat);
  k_raw_emb<<<BT, 512, 0, stream>>>(seq, W_ly2, b_ly2, out);
  k_scan<<<1, 512, 0, stream>>>(seq, Mmat, cvec, b1vec, Whh0_bf, Wih1_bf, Whh1_bf,
                                h_init, hidden_bf);
  k_feat_final<<<64, 256, 0, stream>>>(hidden_bf, Wfin_bf, b_fin, out);
}
